// GRAPHSAGE_70153995813502
// MI455X (gfx1250) — matrix operand order repaired and checked by reading, not compiled, not run on hardware
//
#include <hip/hip_runtime.h>

// ---------------------------------------------------------------------------
// GraphSAGE (gcn aggregator) + MLP head for MI455X (gfx1250, wave32, WMMA)
// GEMMs: bf16 A/B fragments, f32 accumulate, 32x32 tile per wave (4 wmma/K-step)
// ---------------------------------------------------------------------------

typedef __attribute__((ext_vector_type(16))) __bf16          v16bf;
typedef __attribute__((ext_vector_type(8)))  float           v8f;
typedef __attribute__((ext_vector_type(16))) unsigned short  v16u;
typedef __attribute__((ext_vector_type(8)))  unsigned short  v8u;

#define DIM 512          // IN_DIM == HID == 512
#define MLP_HID 200
#define MLP_PAD 224      // padded to 7 column-tile PAIRS of 32

__device__ __forceinline__ unsigned short f2bf_u(float f) {
  // round-to-nearest-even fp32 -> bf16 (payload as u16)
  unsigned u = __builtin_bit_cast(unsigned, f);
  unsigned r = u + 0x7fffu + ((u >> 16) & 1u);
  return (unsigned short)(r >> 16);
}

__device__ __forceinline__ v16bf mk_a(const unsigned short* p, int k0, int half) {
  v8u lo = *(const v8u*)(p + k0 + half * 16);
  v8u hi = *(const v8u*)(p + k0 + half * 16 + 8);
  v16u u = __builtin_shufflevector(lo, hi, 0,1,2,3,4,5,6,7,8,9,10,11,12,13,14,15);
  return __builtin_bit_cast(v16bf, u);
}

// ---------------------------------------------------------------------------
// degree count: deg[dst[e]] += 1
// ---------------------------------------------------------------------------
__global__ void deg_count(const int* __restrict__ dst, float* __restrict__ deg, int e) {
  int i = blockIdx.x * blockDim.x + threadIdx.x;
  if (i < e) atomicAdd(&deg[dst[i]], 1.0f);
}

// ---------------------------------------------------------------------------
// pack W (K x ncols, row-major fp32) into WMMA B-fragment bf16 layout:
//   Bp[ ((k/16)*npad + n)*16 + (k%16) ] = bf16(W[k][n]),  zero-pad n>=ncols
// Lane (half,n) of a wave then loads 16 contiguous bf16 = one 32B load.
// ---------------------------------------------------------------------------
__global__ void pack_b(const float* __restrict__ W, unsigned short* __restrict__ Bp,
                       int K, int ncols, int npad) {
  int idx = blockIdx.x * blockDim.x + threadIdx.x;
  if (idx >= K * npad) return;
  int k = idx / npad, n = idx - k * npad;
  float v = (n < ncols) ? W[(size_t)k * ncols + n] : 0.0f;
  Bp[(((size_t)(k >> 4) * npad + n) << 4) + (k & 15)] = f2bf_u(v);
}

// ---------------------------------------------------------------------------
// edge scatter: agg[dst[e]][:] += feat[src[e]][:]   (one block per edge)
// agg (102MB) is L2-resident on MI455X (192MB L2) -> atomics stay in L2.
// ---------------------------------------------------------------------------
__global__ void scatter_add(const float* __restrict__ feat, const int* __restrict__ src,
                            const int* __restrict__ dst, float* __restrict__ agg) {
  int e = blockIdx.x;
  int s = src[e], d = dst[e];
  const float* from = feat + (size_t)s * DIM;
  float* to = agg + (size_t)d * DIM;
  for (int k = threadIdx.x; k < DIM; k += blockDim.x)
    atomicAdd(&to[k], from[k]);
}

// ---------------------------------------------------------------------------
// combine + bf16 convert: Abf = bf16( (agg + 2*hin) / (deg+2) )
// ---------------------------------------------------------------------------
__global__ void combine_bf16(const float* __restrict__ hin, const float* __restrict__ deg,
                             const float* __restrict__ agg, unsigned short* __restrict__ outb,
                             size_t total) {
  size_t idx = (size_t)blockIdx.x * blockDim.x + threadIdx.x;
  if (idx >= total) return;
  float d = deg[idx >> 9];  // idx / 512
  outb[idx] = f2bf_u((agg[idx] + 2.0f * hin[idx]) / (d + 2.0f));
}

// ---------------------------------------------------------------------------
// WMMA GEMM: C(M x nvalid) = relu(A(M x K, bf16 row-major) @ Bpacked + bias)
// 8 waves/block; each wave owns a 32x32 output tile:
//   2 A fragments x 2 B fragments -> 4 x v_wmma_f32_16x16x32_bf16 per K-step.
// outf (fp32, ld=ldout) and/or outb (bf16, ld=ldout) may be null.
// ---------------------------------------------------------------------------
__global__ void __launch_bounds__(256)
gemm_wmma(const unsigned short* __restrict__ A, const unsigned short* __restrict__ Bp,
          const float* __restrict__ bias, float* __restrict__ outf,
          unsigned short* __restrict__ outb,
          int M, int K, int npad, int nvalid, int ldout, int relu) {
  const int lane = threadIdx.x & 31;
  const int wave = threadIdx.x >> 5;
  const int colpair = blockIdx.y * 8 + wave;
  if (colpair * 32 >= npad) return;

  const int half = lane >> 4;    // 0: lanes 0-15, 1: lanes 16-31
  const int m16  = lane & 15;
  const int rbase = blockIdx.x * 32;
  int ar0 = rbase + m16;       if (ar0 >= M) ar0 = M - 1;   // clamped loads,
  int ar1 = rbase + 16 + m16;  if (ar1 >= M) ar1 = M - 1;   // guarded stores
  const unsigned short* ap0 = A + (size_t)ar0 * K;
  const unsigned short* ap1 = A + (size_t)ar1 * K;
  const int gc0 = colpair * 32 + m16;
  const int gc1 = gc0 + 16;

  v8f acc00 = {}, acc01 = {}, acc10 = {}, acc11 = {};
  const int kblocks = K >> 5;
#pragma unroll 2
  for (int kb = 0; kb < kblocks; ++kb) {
    const int k0 = kb << 5;
    // prefetch the A stream ahead (global_prefetch_b8)
    __builtin_prefetch(ap0 + k0 + 256, 0, 1);
    __builtin_prefetch(ap1 + k0 + 256, 0, 1);
    // A fragments (contiguous k order, matching B): half=0 -> K 0-15, half=1 -> K 16-31
    v16bf a0 = mk_a(ap0, k0, half);
    v16bf a1 = mk_a(ap1, k0, half);
    // B fragments: lane (half, gc) holds B[k0+half*16 + 0..15][gc], packed contiguous
    const size_t boff = ((size_t)(kb * 2 + half) * npad) << 4;
    v16bf b0 = __builtin_bit_cast(v16bf, *(const v16u*)(Bp + boff + ((size_t)gc0 << 4)));
    v16bf b1 = __builtin_bit_cast(v16bf, *(const v16u*)(Bp + boff + ((size_t)gc1 << 4)));
    acc00 = __builtin_amdgcn_wmma_f32_16x16x32_bf16(false, a0, false, b0, (short)0, acc00, false, false);
    acc01 = __builtin_amdgcn_wmma_f32_16x16x32_bf16(false, a0, false, b1, (short)0, acc01, false, false);
    acc10 = __builtin_amdgcn_wmma_f32_16x16x32_bf16(false, a1, false, b0, (short)0, acc10, false, false);
    acc11 = __builtin_amdgcn_wmma_f32_16x16x32_bf16(false, a1, false, b1, (short)0, acc11, false, false);
  }

  auto store_tile = [&](const v8f& acc, int mbase, int gcol) {
    if (gcol >= nvalid) return;
    const float bval = bias[gcol];
#pragma unroll
    for (int r = 0; r < 8; ++r) {
      int m = mbase + r + half * 8;   // C/D layout: VGPR r -> row r + 8*half
      if (m >= M) continue;
      float v = acc[r] + bval;
      if (relu) v = fmaxf(v, 0.0f);
      if (outf) outf[(size_t)m * ldout + gcol] = v;
      if (outb) outb[(size_t)m * ldout + gcol] = f2bf_u(v);
    }
  };
  store_tile(acc00, rbase,      gc0);
  store_tile(acc01, rbase,      gc1);
  store_tile(acc10, rbase + 16, gc0);
  store_tile(acc11, rbase + 16, gc1);
}

// ---------------------------------------------------------------------------
// batchnorm stats (biased var): one block per feature column j of h3 (ld=MLP_HID)
// sc[j] = rsqrt(var+eps)*gamma[j];  tt[j] = beta[j] - mean*sc[j]
// ---------------------------------------------------------------------------
__global__ void bn_stats(const float* __restrict__ h3, const float* __restrict__ gamma,
                         const float* __restrict__ beta, float* __restrict__ sc,
                         float* __restrict__ tt, int n) {
  const int j = blockIdx.x;
  float s = 0.0f, q = 0.0f;
  for (int i = threadIdx.x; i < n; i += blockDim.x) {
    float v = h3[(size_t)i * MLP_HID + j];
    s += v; q += v * v;
  }
  __shared__ float ss[256], qq[256];
  ss[threadIdx.x] = s; qq[threadIdx.x] = q;
  __syncthreads();
  for (int off = 128; off > 0; off >>= 1) {
    if (threadIdx.x < off) { ss[threadIdx.x] += ss[threadIdx.x + off];
                             qq[threadIdx.x] += qq[threadIdx.x + off]; }
    __syncthreads();
  }
  if (threadIdx.x == 0) {
    float mean = ss[0] / n;
    float var  = qq[0] / n - mean * mean;
    float r = rsqrtf(var + 1e-5f);
    float scale = r * gamma[j];
    sc[j] = scale;
    tt[j] = beta[j] - mean * scale;
  }
}

// ---------------------------------------------------------------------------
// fold BN affine into the 2-class projection:
//   WoP[j][c] = sc[j]*Wo[j][c];  cst[c] = sum_j tt[j]*Wo[j][c] + bo[c]
// ---------------------------------------------------------------------------
__global__ void wo_prep(const float* __restrict__ sc, const float* __restrict__ tt,
                        const float* __restrict__ Wo, const float* __restrict__ bo,
                        float* __restrict__ WoP, float* __restrict__ cst) {
  __shared__ float c0[256], c1[256];
  const int j = threadIdx.x;
  float p0 = 0.0f, p1 = 0.0f;
  if (j < MLP_HID) {
    float w0 = Wo[j * 2], w1 = Wo[j * 2 + 1];
    WoP[j * 2]     = sc[j] * w0;
    WoP[j * 2 + 1] = sc[j] * w1;
    p0 = tt[j] * w0; p1 = tt[j] * w1;
  }
  c0[j] = p0; c1[j] = p1;
  __syncthreads();
  for (int off = 128; off > 0; off >>= 1) {
    if (j < off) { c0[j] += c0[j + off]; c1[j] += c1[j + off]; }
    __syncthreads();
  }
  if (j == 0) { cst[0] = c0[0] + bo[0]; cst[1] = c1[0] + bo[1]; }
}

// ---------------------------------------------------------------------------
// final: out[i][c] = dot(h3[i][:], WoP[:,c]) + cst[c]  (thread per row)
// ---------------------------------------------------------------------------
__global__ void final_out(const float* __restrict__ h3, const float* __restrict__ WoP,
                          const float* __restrict__ cst, float* __restrict__ out, int n) {
  int i = blockIdx.x * blockDim.x + threadIdx.x;
  if (i >= n) return;
  const float* row = h3 + (size_t)i * MLP_HID;
  float a0 = cst[0], a1 = cst[1];
#pragma unroll 4
  for (int j = 0; j < MLP_HID; ++j) {
    float v = row[j];
    a0 += v * WoP[2 * j];
    a1 += v * WoP[2 * j + 1];
  }
  out[2 * i] = a0;
  out[2 * i + 1] = a1;
}

// ---------------------------------------------------------------------------
extern "C" void kernel_launch(void* const* d_in, const int* in_sizes, int n_in,
                              void* d_out, int out_size, void* d_ws, size_t ws_size,
                              hipStream_t stream) {
  const float* features = (const float*)d_in[0];
  const int*   src      = (const int*)d_in[1];
  const int*   dst      = (const int*)d_in[2];
  const float* W1 = (const float*)d_in[3];
  const float* b1 = (const float*)d_in[4];
  const float* W2 = (const float*)d_in[5];
  const float* b2 = (const float*)d_in[6];
  const float* Wh = (const float*)d_in[7];
  const float* bh = (const float*)d_in[8];
  const float* gm = (const float*)d_in[9];
  const float* bt = (const float*)d_in[10];
  const float* Wo = (const float*)d_in[11];
  const float* bo = (const float*)d_in[12];

  const int N = in_sizes[0] / DIM;   // 50000
  const int E = in_sizes[1];         // 800000
  const size_t ND = (size_t)N * DIM;

  // workspace carve-out (256B aligned)
  char* base = (char*)d_ws;
  size_t off = 0;
  auto take = [&](size_t bytes) -> char* {
    char* p = base + off;
    off = (off + bytes + 255) & ~(size_t)255;
    return p;
  };
  float*          deg = (float*)take((size_t)N * 4);
  float*          agg = (float*)take(ND * 4);
  float*          h   = (float*)take(ND * 4);
  unsigned short* Abf = (unsigned short*)take(ND * 2);
  unsigned short* Bp1 = (unsigned short*)take((size_t)DIM * DIM * 2);
  unsigned short* Bp2 = (unsigned short*)take((size_t)DIM * DIM * 2);
  unsigned short* BpH = (unsigned short*)take((size_t)DIM * MLP_PAD * 2);
  float*          sc  = (float*)take(MLP_HID * 4);
  float*          tt  = (float*)take(MLP_HID * 4);
  float*          WoP = (float*)take(MLP_HID * 2 * 4);
  float*          cst = (float*)take(2 * 4);
  // overlays (lifetimes disjoint): layer2 bf16 output reuses h; h3 reuses agg
  unsigned short* Abf2 = (unsigned short*)h;
  float*          h3   = agg;

  // degrees + weight packing (independent of graph passes)
  hipMemsetAsync(deg, 0, (size_t)N * 4, stream);
  hipMemsetAsync(agg, 0, ND * 4, stream);
  deg_count<<<(E + 255) / 256, 256, 0, stream>>>(dst, deg, E);
  pack_b<<<(DIM * DIM + 255) / 256, 256, 0, stream>>>(W1, Bp1, DIM, DIM, DIM);
  pack_b<<<(DIM * DIM + 255) / 256, 256, 0, stream>>>(W2, Bp2, DIM, DIM, DIM);
  pack_b<<<(DIM * MLP_PAD + 255) / 256, 256, 0, stream>>>(Wh, BpH, DIM, MLP_HID, MLP_PAD);

  const unsigned cgrid = (unsigned)((ND + 255) / 256);
  // 32 rows x 256 cols per block (8 waves x 32x32 tiles)
  dim3 g512((N + 31) / 32, (DIM / 32 + 7) / 8);       // 1563 x 2
  dim3 gmlp((N + 31) / 32, (MLP_PAD / 32 + 7) / 8);   // 1563 x 1

  // ---- SAGE layer 1 ----
  scatter_add<<<E, 256, 0, stream>>>(features, src, dst, agg);
  combine_bf16<<<cgrid, 256, 0, stream>>>(features, deg, agg, Abf, ND);
  gemm_wmma<<<g512, 256, 0, stream>>>(Abf, Bp1, b1, h, nullptr,
                                      N, DIM, DIM, DIM, DIM, 1);

  // ---- SAGE layer 2 ----
  hipMemsetAsync(agg, 0, ND * 4, stream);
  scatter_add<<<E, 256, 0, stream>>>(h, src, dst, agg);
  combine_bf16<<<cgrid, 256, 0, stream>>>(h, deg, agg, Abf, ND);
  gemm_wmma<<<g512, 256, 0, stream>>>(Abf, Bp2, b2, nullptr, Abf2,
                                      N, DIM, DIM, DIM, DIM, 1);

  // ---- MLP hidden: h3 = relu(h2 @ Wh + bh) ----
  gemm_wmma<<<gmlp, 256, 0, stream>>>(Abf2, BpH, bh, h3, nullptr,
                                      N, DIM, MLP_PAD, MLP_HID, MLP_HID, 1);

  // ---- batchnorm + folded 200->2 projection ----
  bn_stats<<<MLP_HID, 256, 0, stream>>>(h3, gm, bt, sc, tt, N);
  wo_prep<<<1, 256, 0, stream>>>(sc, tt, Wo, bo, WoP, cst);
  final_out<<<(N + 255) / 256, 256, 0, stream>>>(h3, WoP, cst, (float*)d_out, N);
}
